// LearnableInterpolator_53884659695694
// MI455X (gfx1250) — matrix operand order repaired and checked by reading, not compiled, not run on hardware
//
#include <hip/hip_runtime.h>
#include <hip/hip_bf16.h>

typedef __attribute__((ext_vector_type(16))) _Float16 v16h;
typedef __attribute__((ext_vector_type(8)))  _Float16 v8h;
typedef __attribute__((ext_vector_type(8)))  float    v8f;

// Problem dims
#define BD   2
#define DD   128
#define HD   64
#define WD   96
#define HU   128
#define WU   192
#define NPIX (BD*HD*WD)        // 12288 base pixels
#define NTOK (BD*HU*WU)        // 49152 tokens

// Workspace layout (units: _Float16 halves)
#define O_WQ   ((size_t)0)                 // 2*128*128
#define O_WO   ((size_t)32768)             // 2*128*128
#define O_WKV  ((size_t)65536)             // 2*256*128  ([N=256][K=128] per layer)
#define O_W1   ((size_t)131072)            // 2*512*128
#define O_W2   ((size_t)262144)            // 2*128*512
#define O_WQF  ((size_t)393216)            // 128*128
#define O_WCF  ((size_t)409600)            // 128*128
#define O_K    ((size_t)425984)            // 2*12288*128
#define O_V    ((size_t)3571712)           // 2*12288*128
#define O_CP   ((size_t)6717440)           // 12288*128
// total 8290304 halves = ~16.6 MB

// ---------------- WMMA fragment helpers (CDNA5 16x16x32 f16) ----------------

__device__ inline v16h load_A_frag(const _Float16* aw, int ldA, int k0, int lane) {
    // A 16x32 f16: lanes 0-15 row M, elems 0..15 = K(k0+0..15)
    //              lanes 16-31 same rows, K(k0+16..31).
    int m  = lane & 15;
    int kh = (lane >> 4) << 4;
    const _Float16* p = aw + m * ldA + k0 + kh;
    union { v16h v; v8h h[2]; } u;
    u.h[0] = *(const v8h*)(p);
    u.h[1] = *(const v8h*)(p + 8);
    return u.v;
}

__device__ inline v16h load_B_frag(const _Float16* wt, int Kd, int n0, int k0, int lane) {
    // B 32x16 f16 from [N][K] weight: lanes 0-15 col N=n0+lane, K=k0..k0+15;
    // lanes 16-31 col N=n0+lane-16, K=k0+16..k0+31. 16 contiguous halves.
    int n  = lane & 15;
    int kb = (lane >> 4) << 4;
    return *(const v16h*)(wt + (size_t)(n0 + n) * Kd + k0 + kb);
}

__device__ inline v8f wmma_mm(const _Float16* aw, int ldA, int kA0,
                              const _Float16* wt, int Kd, int n0, int kB0,
                              int nsteps, float bias, int lane) {
    v8f c;
#pragma unroll
    for (int r = 0; r < 8; ++r) c[r] = bias;
    for (int ks = 0; ks < nsteps; ++ks) {
        v16h a = load_A_frag(aw, ldA, kA0 + ks * 32, lane);
        v16h b = load_B_frag(wt, Kd, n0, kB0 + ks * 32, lane);
        c = __builtin_amdgcn_wmma_f32_16x16x32_f16(false, a, false, b, (short)0, c, false, false);
    }
    return c;
}

// ---------------- small math helpers ----------------

__device__ inline float gelu_f(float x) {
    float x3 = x * x * x;
    return 0.5f * x * (1.0f + tanhf(0.7978845608028654f * (x + 0.044715f * x3)));
}

// LN across 128 channels; 2 lanes per token (lane&15 = token, lane>>4 = half).
// In-place variant (stage A).
__device__ inline void wave_ln_inplace(float* xr, float* red, int lane) {
    int m = lane & 15, hf = lane >> 4;
    float* p = xr + m * 128 + hf * 64;
    float s1 = 0.f, s2 = 0.f;
    for (int c = 0; c < 64; ++c) { float v = p[c]; s1 += v; s2 += v * v; }
    red[lane] = s1; red[lane + 32] = s2;
    __syncthreads();
    float s1t = s1 + red[lane ^ 16];
    float s2t = s2 + red[(lane ^ 16) + 32];
    float mu   = s1t * (1.f / 128.f);
    float var  = s2t * (1.f / 128.f) - mu * mu;
    float rstd = rsqrtf(var + 1e-6f);
    for (int c = 0; c < 64; ++c) p[c] = (p[c] - mu) * rstd;
    __syncthreads();
}

// Non-destructive variant: writes f16 into ar (stage B).
__device__ inline void wave_ln_to_f16(const float* xr, _Float16* ar, float* red, int lane) {
    int m = lane & 15, hf = lane >> 4;
    const float* p = xr + m * 128 + hf * 64;
    float s1 = 0.f, s2 = 0.f;
    for (int c = 0; c < 64; ++c) { float v = p[c]; s1 += v; s2 += v * v; }
    red[lane] = s1; red[lane + 32] = s2;
    __syncthreads();
    float s1t = s1 + red[lane ^ 16];
    float s2t = s2 + red[(lane ^ 16) + 32];
    float mu   = s1t * (1.f / 128.f);
    float var  = s2t * (1.f / 128.f) - mu * mu;
    float rstd = rsqrtf(var + 1e-6f);
    _Float16* q = ar + m * 128 + hf * 64;
    for (int c = 0; c < 64; ++c) q[c] = (_Float16)((p[c] - mu) * rstd);
    __syncthreads();
}

// ---------------- kernel 0: weight transpose fp32[K][N] -> fp16[N][K] ----------------

__global__ __launch_bounds__(256) void wconv_t_f16(const float* __restrict__ src,
                                                   _Float16* __restrict__ dst,
                                                   int Kd, int Nd) {
    int i = blockIdx.x * blockDim.x + threadIdx.x;
    if (i >= Kd * Nd) return;
    int k = i / Nd, n = i % Nd;
    dst[(size_t)n * Kd + k] = (_Float16)src[i];
}

// ---------------- stage A: per-base-pixel LN + K/V/CTXP projections ----------------

__global__ __launch_bounds__(128) void stage_a(
    const float* __restrict__ feat_map,
    const float* __restrict__ nc_w, const float* __restrict__ nc_b,
    const float* __restrict__ bkv,  const float* __restrict__ bcf,
    const _Float16* __restrict__ wkv_t, const _Float16* __restrict__ wcf_t,
    _Float16* __restrict__ kbuf, _Float16* __restrict__ vbuf, _Float16* __restrict__ cpbuf) {

    __shared__ float    sxf[4][16 * 128];
    __shared__ _Float16 sah[4][16 * 128];
    __shared__ float    sred[4][64];

    int wave = threadIdx.x >> 5, lane = threadIdx.x & 31;
    float*    xf  = sxf[wave];
    _Float16* ah  = sah[wave];
    float*    red = sred[wave];

    int tile = blockIdx.x * 4 + wave;          // 768 tiles of 16 pixels
    int pix0 = tile * 16;
    int w0 = pix0 % WD;
    int h  = (pix0 / WD) % HD;
    int b  = pix0 / (WD * HD);
    int m = lane & 15, hf = lane >> 4;

    // gather 16 pixels x 128 channels (channel-major source)
    for (int c = 0; c < 128; c += 2) {
        int cc = c + hf;
        xf[m * 128 + cc] = feat_map[(((size_t)b * 128 + cc) * HD + h) * WD + w0 + m];
    }
    __syncthreads();

    wave_ln_inplace(xf, red, lane);            // xf now holds LN(pixel)

    for (int l = 0; l < 2; ++l) {
        // ctx_n = ln * nc_w[l] + nc_b[l]  ->  f16 A tile
        for (int c = 0; c < 64; ++c) {
            int cc = hf * 64 + c;
            ah[m * 128 + cc] =
                (_Float16)(xf[m * 128 + cc] * nc_w[l * 128 + cc] + nc_b[l * 128 + cc]);
        }
        __syncthreads();
        // kv = ctx_n @ wkv[l] + bkv[l]   (K=128, N=256)
        for (int nt = 0; nt < 16; ++nt) {
            int n0 = nt * 16;
            int n  = n0 + (lane & 15);
            v8f cacc = wmma_mm(ah, 128, 0, wkv_t + (size_t)l * 256 * 128, 128, n0, 0,
                               4, bkv[l * 256 + n], lane);
#pragma unroll
            for (int r = 0; r < 8; ++r) {
                int row = ((lane >> 4) << 3) + r;
                int pix = pix0 + row;
                _Float16 v = (_Float16)cacc[r];
                if (n < 128) kbuf[((size_t)l * NPIX + pix) * 128 + n]         = v;
                else         vbuf[((size_t)l * NPIX + pix) * 128 + (n - 128)] = v;
            }
        }
        __syncthreads();
    }

    // ctxp = ln @ wcf + bcf
    for (int c = 0; c < 64; ++c) {
        int cc = hf * 64 + c;
        ah[m * 128 + cc] = (_Float16)xf[m * 128 + cc];
    }
    __syncthreads();
    for (int nt = 0; nt < 8; ++nt) {
        int n0 = nt * 16;
        int n  = n0 + (lane & 15);
        v8f cacc = wmma_mm(ah, 128, 0, wcf_t, 128, n0, 0, 4, bcf[n], lane);
#pragma unroll
        for (int r = 0; r < 8; ++r) {
            int row = ((lane >> 4) << 3) + r;
            cpbuf[((size_t)(pix0 + row)) * 128 + n] = (_Float16)cacc[r];
        }
    }
}

// ---------------- stage B: per-token transformer + final mask ----------------

#define WAVE_BYTES 30080   // xf 8192 | qb 8192 | ah 4096 | hb 8192 | red 256 | plog 1152

__global__ __launch_bounds__(128) void stage_b(
    const float* __restrict__ feat_up,
    const float* __restrict__ bq, const float* __restrict__ bo,
    const float* __restrict__ b1, const float* __restrict__ b2,
    const float* __restrict__ bqf,
    const _Float16* __restrict__ wq_t, const _Float16* __restrict__ wo_t,
    const _Float16* __restrict__ w1_t, const _Float16* __restrict__ w2_t,
    const _Float16* __restrict__ wqf_t,
    const _Float16* __restrict__ kbuf, const _Float16* __restrict__ vbuf,
    const _Float16* __restrict__ cpbuf,
    float* __restrict__ out) {

    extern __shared__ char smem[];
    int wave = threadIdx.x >> 5, lane = threadIdx.x & 31;
    char* base = smem + (size_t)wave * WAVE_BYTES;
    float*    xf   = (float*)(base);
    float*    qb   = (float*)(base + 8192);
    _Float16* ah   = (_Float16*)(base + 16384);
    _Float16* hb   = (_Float16*)(base + 20480);
    float*    red  = (float*)(base + 28672);
    float*    plog = (float*)(base + 28928);

    int tile = blockIdx.x * 4 + wave;          // 3072 tiles of 16 tokens
    int t0  = tile * 16;
    int wu0 = t0 % WU;
    int hu  = (t0 / WU) % HU;
    int b   = t0 / (WU * HU);
    int m = lane & 15, hf = lane >> 4;
    int c0 = hf * 64;

    // 3x3 edge-clamped neighbor pixel indices for this lane's token
    int h = hu >> 1;
    int wpix = (wu0 + m) >> 1;
    int pj9[9];
#pragma unroll
    for (int iy = 0; iy < 3; ++iy) {
#pragma unroll
        for (int ix = 0; ix < 3; ++ix) {
            int hh = h - 1 + iy; hh = hh < 0 ? 0 : (hh > HD - 1 ? HD - 1 : hh);
            int ww = wpix - 1 + ix; ww = ww < 0 ? 0 : (ww > WD - 1 ? WD - 1 : ww);
            pj9[iy * 3 + ix] = (b * HD + hh) * WD + ww;
        }
    }

    // load residual x (16 tokens x 128 channels) from feat_map_up
    for (int c = 0; c < 128; c += 2) {
        int cc = c + hf;
        xf[m * 128 + cc] = feat_up[(((size_t)b * 128 + cc) * HU + hu) * WU + wu0 + m];
    }
    __syncthreads();

    const float ascale = 0.17677669529663687f;   // 1/sqrt(32)

    for (int l = 0; l < 2; ++l) {
        // ---- q = LN(x) @ wq[l] + bq[l] ----
        wave_ln_to_f16(xf, ah, red, lane);
        for (int nt = 0; nt < 8; ++nt) {
            int n0 = nt * 16;
            int n  = n0 + (lane & 15);
            v8f cacc = wmma_mm(ah, 128, 0, wq_t + (size_t)l * 16384, 128, n0, 0,
                               4, bq[l * 128 + n], lane);
#pragma unroll
            for (int r = 0; r < 8; ++r) {
                int row = ((lane >> 4) << 3) + r;
                qb[row * 128 + n] = cacc[r];
            }
        }
        __syncthreads();

        // ---- 9-neighbor attention (2 lanes/token, 2 heads each) ----
        const size_t kvbase = (size_t)l * NPIX * 128;
        const _Float16* vp[9];
        float sA[9], sB[9];
#pragma unroll
        for (int j = 0; j < 9; ++j) {
            const _Float16* kp = kbuf + kvbase + (size_t)pj9[j] * 128 + c0;
            vp[j]              = vbuf + kvbase + (size_t)pj9[j] * 128 + c0;
            const float* qp = qb + m * 128 + c0;
            float a0 = 0.f, a1 = 0.f;
            for (int c = 0; c < 32; ++c) {
                a0 += qp[c]      * (float)kp[c];
                a1 += qp[32 + c] * (float)kp[32 + c];
            }
            sA[j] = a0 * ascale; sB[j] = a1 * ascale;
        }
        float mA = sA[0], mB = sB[0];
#pragma unroll
        for (int j = 1; j < 9; ++j) { mA = fmaxf(mA, sA[j]); mB = fmaxf(mB, sB[j]); }
        float dA = 0.f, dB = 0.f;
#pragma unroll
        for (int j = 0; j < 9; ++j) {
            sA[j] = expf(sA[j] - mA); dA += sA[j];
            sB[j] = expf(sB[j] - mB); dB += sB[j];
        }
        float rA = 1.f / dA, rB = 1.f / dB;
#pragma unroll
        for (int j = 0; j < 9; ++j) { sA[j] *= rA; sB[j] *= rB; }
        for (int c = 0; c < 32; ++c) {
            float oa = 0.f, ob = 0.f;
#pragma unroll
            for (int j = 0; j < 9; ++j) {
                oa += sA[j] * (float)vp[j][c];
                ob += sB[j] * (float)vp[j][32 + c];
            }
            ah[m * 128 + c0 + c]      = (_Float16)oa;
            ah[m * 128 + c0 + 32 + c] = (_Float16)ob;
        }
        __syncthreads();

        // ---- x += o @ wo[l] + bo[l] ----
        for (int nt = 0; nt < 8; ++nt) {
            int n0 = nt * 16;
            int n  = n0 + (lane & 15);
            v8f cacc = wmma_mm(ah, 128, 0, wo_t + (size_t)l * 16384, 128, n0, 0,
                               4, bo[l * 128 + n], lane);
#pragma unroll
            for (int r = 0; r < 8; ++r) {
                int row = ((lane >> 4) << 3) + r;
                xf[row * 128 + n] += cacc[r];
            }
        }
        __syncthreads();

        // ---- MLP: x += gelu(LN(x) @ w1 + b1) @ w2 + b2 (N=512 split in two) ----
        wave_ln_to_f16(xf, ah, red, lane);
        for (int part = 0; part < 2; ++part) {
            for (int nt = 0; nt < 16; ++nt) {
                int n0g = part * 256 + nt * 16;
                int ng  = n0g + (lane & 15);
                v8f cacc = wmma_mm(ah, 128, 0, w1_t + (size_t)l * 512 * 128, 128, n0g, 0,
                                   4, b1[l * 512 + ng], lane);
#pragma unroll
                for (int r = 0; r < 8; ++r) {
                    int row = ((lane >> 4) << 3) + r;
                    hb[row * 256 + nt * 16 + (lane & 15)] = (_Float16)gelu_f(cacc[r]);
                }
            }
            __syncthreads();
            for (int nt = 0; nt < 8; ++nt) {
                int n0 = nt * 16;
                int n  = n0 + (lane & 15);
                v8f cacc = wmma_mm(hb, 256, 0, w2_t + (size_t)l * 128 * 512, 512, n0,
                                   part * 256, 8, part == 0 ? b2[l * 128 + n] : 0.f, lane);
#pragma unroll
                for (int r = 0; r < 8; ++r) {
                    int row = ((lane >> 4) << 3) + r;
                    xf[row * 128 + n] += cacc[r];
                }
            }
            __syncthreads();
        }
    }

    // ---- final: xq = LN(x) @ wqf + bqf; logits vs ctxp; softmax ----
    wave_ln_to_f16(xf, ah, red, lane);
    for (int nt = 0; nt < 8; ++nt) {
        int n0 = nt * 16;
        int n  = n0 + (lane & 15);
        v8f cacc = wmma_mm(ah, 128, 0, wqf_t, 128, n0, 0, 4, bqf[n], lane);
#pragma unroll
        for (int r = 0; r < 8; ++r) {
            int row = ((lane >> 4) << 3) + r;
            qb[row * 128 + n] = cacc[r];
        }
    }
    __syncthreads();

#pragma unroll
    for (int j = 0; j < 9; ++j) {
        const _Float16* cp = cpbuf + (size_t)pj9[j] * 128 + c0;
        const float* qp = qb + m * 128 + c0;
        float a = 0.f;
        for (int c = 0; c < 64; ++c) a += qp[c] * (float)cp[c];
        plog[m * 18 + hf * 9 + j] = a;
    }
    __syncthreads();

    if (lane < 16) {
        int mm = lane;
        float lg[9];
#pragma unroll
        for (int j = 0; j < 9; ++j)
            lg[j] = (plog[mm * 18 + j] + plog[mm * 18 + 9 + j]) * 0.08838834764831845f;
        float mx = lg[0];
#pragma unroll
        for (int j = 1; j < 9; ++j) mx = fmaxf(mx, lg[j]);
        float dn = 0.f;
#pragma unroll
        for (int j = 0; j < 9; ++j) { lg[j] = expf(lg[j] - mx); dn += lg[j]; }
        float rdn = 1.f / dn;
        int wu = wu0 + mm;
#pragma unroll
        for (int j = 0; j < 9; ++j)
            out[(((size_t)b * 9 + j) * HU + hu) * WU + wu] = lg[j] * rdn;
    }
}

// ---------------- host launcher ----------------

extern "C" void kernel_launch(void* const* d_in, const int* in_sizes, int n_in,
                              void* d_out, int out_size, void* d_ws, size_t ws_size,
                              hipStream_t stream) {
    (void)in_sizes; (void)n_in; (void)out_size; (void)ws_size;

    const float* feat_map = (const float*)d_in[0];
    const float* feat_up  = (const float*)d_in[1];
    const float* nc_w = (const float*)d_in[2];
    const float* nc_b = (const float*)d_in[3];
    const float* wq   = (const float*)d_in[4];
    const float* bq   = (const float*)d_in[5];
    const float* wkv  = (const float*)d_in[6];
    const float* bkv  = (const float*)d_in[7];
    const float* wo   = (const float*)d_in[8];
    const float* bo   = (const float*)d_in[9];
    const float* w1   = (const float*)d_in[10];
    const float* b1   = (const float*)d_in[11];
    const float* w2   = (const float*)d_in[12];
    const float* b2   = (const float*)d_in[13];
    const float* wqf  = (const float*)d_in[14];
    const float* bqf  = (const float*)d_in[15];
    const float* wcf  = (const float*)d_in[16];
    const float* bcf  = (const float*)d_in[17];
    float* out = (float*)d_out;

    _Float16* ws = (_Float16*)d_ws;

    // weight conversion: fp32 [K][N] -> fp16 [N][K]
    for (int l = 0; l < 2; ++l) {
        wconv_t_f16<<<64,  256, 0, stream>>>(wq  + (size_t)l * 16384, ws + O_WQ  + (size_t)l * 16384, 128, 128);
        wconv_t_f16<<<64,  256, 0, stream>>>(wo  + (size_t)l * 16384, ws + O_WO  + (size_t)l * 16384, 128, 128);
        wconv_t_f16<<<128, 256, 0, stream>>>(wkv + (size_t)l * 32768, ws + O_WKV + (size_t)l * 32768, 128, 256);
        wconv_t_f16<<<256, 256, 0, stream>>>(w1  + (size_t)l * 65536, ws + O_W1  + (size_t)l * 65536, 128, 512);
        wconv_t_f16<<<256, 256, 0, stream>>>(w2  + (size_t)l * 65536, ws + O_W2  + (size_t)l * 65536, 512, 128);
    }
    wconv_t_f16<<<64, 256, 0, stream>>>(wqf, ws + O_WQF, 128, 128);
    wconv_t_f16<<<64, 256, 0, stream>>>(wcf, ws + O_WCF, 128, 128);

    // stage A: 12288 base pixels, 16 per wave, 4 waves per block
    stage_a<<<NPIX / 64, 128, 0, stream>>>(
        feat_map, nc_w, nc_b, bkv, bcf,
        ws + O_WKV, ws + O_WCF,
        ws + O_K, ws + O_V, ws + O_CP);

    // stage B: 49152 tokens, 16 per wave, 4 waves per block, 120320 B dynamic LDS
    stage_b<<<NTOK / 64, 128, 4 * WAVE_BYTES, stream>>>(
        feat_up, bq, bo, b1, b2, bqf,
        ws + O_WQ, ws + O_WO, ws + O_W1, ws + O_W2, ws + O_WQF,
        ws + O_K, ws + O_V, ws + O_CP,
        out);
}
